// StudentTCodebook_22419729285664
// MI455X (gfx1250) — matrix operand order repaired and checked by reading, not compiled, not run on hardware
//
#include <hip/hip_runtime.h>
#include <stdint.h>

typedef __bf16 bf16;
typedef __attribute__((ext_vector_type(16))) __bf16 v16bf;
typedef __attribute__((ext_vector_type(8)))  float  v8f;
typedef __attribute__((ext_vector_type(4)))  unsigned int u32x4;
typedef __attribute__((ext_vector_type(4)))  float  f32x4;

#define DIM     256   // D
#define TILE_M  64    // rows of x per block
#define XPITCH  264   // halves per x row in LDS (528B pitch -> conflict-free A-frag b128 loads)

// ---------- fast path (workspace-backed) ----------
#define CH      32    // codes per K-chunk
#define EP      40    // padded pitch (halves) of transposed panel rows (80B -> conflict-free B-frag)
#define PANEL_HALVES (2*DIM*EP)        // hi panel + lo panel = 20480 halves
#define PANEL_BYTES  (PANEL_HALVES*2)  // 40960 B per chunk
#define MAIN_SMEM (2*(TILE_M*XPITCH*2) + 2*PANEL_BYTES + TILE_M*4)   // 149760 B

// ---------- fallback path (no workspace) ----------
#define FCH     64
#define FEP     72
#define FB_SMEM (2*(TILE_M*XPITCH*2) + 2*(DIM*FEP*2) + FCH*4 + TILE_M*4)

__device__ __forceinline__ void split_bf16(float f, bf16& hi, bf16& lo) {
    hi = (bf16)f;
    lo = (bf16)(f - (float)hi);
}

union Frag { v16bf v; u32x4 q[2]; };

__device__ __forceinline__ v8f wmma_bf16(v16bf a, v16bf b, v8f c) {
    return __builtin_amdgcn_wmma_f32_16x16x32_bf16(false, a, false, b, (short)0, c,
                                                   false, false);
}

// Issue the async DMA of one 40960B panel (hi+lo) into LDS at byte offset ldsOff.
// 128 threads x 16B x 20 iterations; per-wave ASYNCcnt tracks completion.
__device__ __forceinline__ void async_fill_panel(const bf16* gsrc, uint32_t ldsOff, int t) {
    unsigned long long g = (unsigned long long)(uintptr_t)gsrc + (uint32_t)t * 16u;
    uint32_t l = ldsOff + (uint32_t)t * 16u;
    #pragma unroll
    for (int r = 0; r < PANEL_BYTES / (128 * 16); ++r) {   // 20
        asm volatile("global_load_async_to_lds_b128 %0, %1, off"
                     :: "v"(l), "v"(g) : "memory");
        g += 128 * 16;
        l += 128 * 16;
    }
}

// ============================================================================
// Pre-kernel: embed f32 [K][D]  ->  per-chunk transposed bf16 hi/lo panels
// panel layout per chunk c: hi[d][0..CH) pitch EP, then lo[d][0..CH) pitch EP.
// Also writes esq_g[k] = 0.5*||e_k||^2.
// ============================================================================
__global__ __launch_bounds__(256)
void vq_pre(const float* __restrict__ embed, bf16* __restrict__ panels,
            float* __restrict__ esq_g)
{
    const int c     = blockIdx.x;
    const int kbase = c * CH;
    const int t     = threadIdx.x;
    const int code  = t >> 3;          // 0..31
    const int dseg  = (t & 7) * 32;    // 8 threads cover D=256

    const f32x4* src = (const f32x4*)(embed + (size_t)(kbase + code) * DIM + dseg);
    bf16* ph = panels + (size_t)c * PANEL_HALVES;     // hi panel
    bf16* pl = ph + DIM * EP;                          // lo panel

    float ssum = 0.0f;
    #pragma unroll
    for (int j = 0; j < 8; ++j) {
        f32x4 v = src[j];
        const int d = dseg + 4 * j;
        #pragma unroll
        for (int i = 0; i < 4; ++i) {
            float f = v[i];
            bf16 hi, lo; split_bf16(f, hi, lo);
            ph[(d + i) * EP + code] = hi;
            pl[(d + i) * EP + code] = lo;
            ssum = fmaf(f, f, ssum);
        }
    }
    ssum += __shfl_xor(ssum, 1);
    ssum += __shfl_xor(ssum, 2);
    ssum += __shfl_xor(ssum, 4);
    if ((t & 7) == 0) esq_g[kbase + code] = 0.5f * ssum;
}

// ============================================================================
// Main kernel: 64 rows/block, async double-buffered panel stream, bf16-split
// WMMA (hi*hi + hi*lo + lo*hi), running per-lane argmax.
// ============================================================================
__global__ __launch_bounds__(128)
void vq_main(const float* __restrict__ x, const bf16* __restrict__ panels,
             const float* __restrict__ esq_g, const float* __restrict__ embed,
             float* __restrict__ out_q, int* __restrict__ out_idx, int K)
{
    extern __shared__ __align__(16) char smem[];
    bf16* xh   = (bf16*)smem;                    // [TILE_M][XPITCH]
    bf16* xl   = xh + TILE_M * XPITCH;
    bf16* ebuf = xl + TILE_M * XPITCH;           // 2 x PANEL_HALVES
    int*  sidx = (int*)(ebuf + 2 * PANEL_HALVES);

    const int t    = threadIdx.x;
    const int lane = t & 31;
    const int wave = t >> 5;
    const int row0 = blockIdx.x * TILE_M;
    // Generic LDS pointers carry the LDS byte offset in their low 32 bits.
    const uint32_t ebase = (uint32_t)(uintptr_t)ebuf;

    const int nChunks = K / CH;

    // Kick off DMA of chunk 0 into buffer 0; it overlaps the x-tile staging.
    async_fill_panel(panels, ebase, t);

    // ---- Stage x tile, split f32 -> bf16 hi/lo into LDS ----
    {
        const int r = t >> 1;
        const int h = t & 1;
        const f32x4* src = (const f32x4*)(x + (size_t)(row0 + r) * DIM + h * 128);
        bf16* dh = xh + r * XPITCH + h * 128;
        bf16* dl = xl + r * XPITCH + h * 128;
        #pragma unroll
        for (int j = 0; j < 32; ++j) {
            f32x4 v = src[j];
            #pragma unroll
            for (int i = 0; i < 4; ++i) {
                bf16 hi, lo; split_bf16(v[i], hi, lo);
                dh[4*j + i] = hi;
                dl[4*j + i] = lo;
            }
        }
    }

    float bestv[8];
    int   besti[8];
    #pragma unroll
    for (int r = 0; r < 8; ++r) { bestv[r] = -__builtin_inff(); besti[r] = 0; }

    for (int c = 0; c < nChunks; ++c) {
        const int kbase = c * CH;

        // Our async ops (current buffer) done, then block-wide handoff.
        asm volatile("s_wait_asynccnt 0x0" ::: "memory");
        __syncthreads();

        // Start DMA of the next chunk into the other buffer; overlaps compute.
        if (c + 1 < nChunks)
            async_fill_panel(panels + (size_t)(c + 1) * PANEL_HALVES,
                             ebase + (uint32_t)((c + 1) & 1) * PANEL_BYTES, t);

        const bf16* eh = ebuf + (size_t)(c & 1) * PANEL_HALVES;
        const bf16* el = eh + DIM * EP;

        v8f acc[2];
        #pragma unroll
        for (int ct = 0; ct < 2; ++ct)
            acc[ct] = (v8f){0.f,0.f,0.f,0.f,0.f,0.f,0.f,0.f};

        const int rowl  = wave * 16 + (lane & 15);
        const int khalf = lane >> 4;

        #pragma unroll
        for (int s = 0; s < 8; ++s) {
            // A fragment (ISA 16-bit A 16x32): halves 0..7 at d0, 8..15 at d0+16
            const int d0 = 32 * s + khalf * 8;
            Frag Ah, Al;
            const u32x4* pah = (const u32x4*)(xh + rowl * XPITCH + d0);
            const u32x4* pal = (const u32x4*)(xl + rowl * XPITCH + d0);
            Ah.q[0] = pah[0]; Ah.q[1] = pah[2];
            Al.q[0] = pal[0]; Al.q[1] = pal[2];

            // B fragment: lane = N column, halves 0..7 at K dB.., 8..15 at dB+16..
            const int dB = 32 * s + khalf * 8;
            #pragma unroll
            for (int ct = 0; ct < 2; ++ct) {
                Frag Bh, Bl;
                #pragma unroll
                for (int e = 0; e < 16; ++e) {
                    const int kk = dB + (e & 7) + (e >> 3) * 16;
                    Bh.v[e] = eh[kk * EP + ct * 16 + (lane & 15)];
                    Bl.v[e] = el[kk * EP + ct * 16 + (lane & 15)];
                }

                acc[ct] = wmma_bf16(Ah.v, Bh.v, acc[ct]);
                acc[ct] = wmma_bf16(Ah.v, Bl.v, acc[ct]);
                acc[ct] = wmma_bf16(Al.v, Bh.v, acc[ct]);
            }
        }

        // Running argmax of score = dot - 0.5*||e||^2 (esq precomputed, L2-hot)
        #pragma unroll
        for (int ct = 0; ct < 2; ++ct) {
            const int nl = ct * 16 + (lane & 15);
            const float hs = esq_g[kbase + nl];
            const int kg = kbase + nl;
            #pragma unroll
            for (int r = 0; r < 8; ++r) {
                const float sc = acc[ct][r] - hs;
                if (sc > bestv[r]) { bestv[r] = sc; besti[r] = kg; }
            }
        }
    }

    // ---- Cross-lane argmax within each 16-lane half (min-index tie-break) ----
    #pragma unroll
    for (int r = 0; r < 8; ++r) {
        float v = bestv[r];
        int   i = besti[r];
        #pragma unroll
        for (int m = 1; m < 16; m <<= 1) {
            const float ov = __shfl_xor(v, m);
            const int   oi = __shfl_xor(i, m);
            if (ov > v || (ov == v && oi < i)) { v = ov; i = oi; }
        }
        if ((lane & 15) == 0)
            sidx[wave * 16 + (lane >> 4) * 8 + r] = i;
    }
    __syncthreads();

    if (t < TILE_M) out_idx[row0 + t] = sidx[t];
    {
        const int r = t >> 1;
        const int h = t & 1;
        const int k = sidx[r];
        const f32x4* sp = (const f32x4*)(embed + (size_t)k * DIM + h * 128);
        f32x4*       dp = (f32x4*)(out_q + (size_t)(row0 + r) * DIM + h * 128);
        #pragma unroll
        for (int j = 0; j < 32; ++j) dp[j] = sp[j];
    }
}

// ============================================================================
// Fallback (no workspace): self-contained version (round-1 kernel, verbatim).
// ============================================================================
__global__ __launch_bounds__(128)
void vq_fallback(const float* __restrict__ x, const float* __restrict__ embed,
                 float* __restrict__ out_q, int* __restrict__ out_idx, int K)
{
    extern __shared__ __align__(16) char smem[];
    bf16*  xh   = (bf16*)smem;
    bf16*  xl   = xh + TILE_M * XPITCH;
    bf16*  eh   = xl + TILE_M * XPITCH;
    bf16*  el   = eh + DIM * FEP;
    float* esq  = (float*)(el + DIM * FEP);
    int*   sidx = (int*)(esq + FCH);

    const int t    = threadIdx.x;
    const int lane = t & 31;
    const int wave = t >> 5;
    const int row0 = blockIdx.x * TILE_M;

    {
        const int r = t >> 1;
        const int h = t & 1;
        const f32x4* src = (const f32x4*)(x + (size_t)(row0 + r) * DIM + h * 128);
        bf16* dh = xh + r * XPITCH + h * 128;
        bf16* dl = xl + r * XPITCH + h * 128;
        #pragma unroll
        for (int j = 0; j < 32; ++j) {
            f32x4 v = src[j];
            #pragma unroll
            for (int i = 0; i < 4; ++i) {
                bf16 hi, lo; split_bf16(v[i], hi, lo);
                dh[4*j + i] = hi;
                dl[4*j + i] = lo;
            }
        }
    }

    float bestv[8];
    int   besti[8];
    #pragma unroll
    for (int r = 0; r < 8; ++r) { bestv[r] = -__builtin_inff(); besti[r] = 0; }

    const int nChunks = K / FCH;
    for (int c = 0; c < nChunks; ++c) {
        const int kbase = c * FCH;
        __syncthreads();
        {
            const int code = t >> 1;
            const int h    = t & 1;
            const f32x4* src = (const f32x4*)(embed + (size_t)(kbase + code) * DIM + h * 128);
            float ssum = 0.0f;
            #pragma unroll
            for (int j = 0; j < 32; ++j) {
                f32x4 v = src[j];
                const int d = h * 128 + 4 * j;
                #pragma unroll
                for (int i = 0; i < 4; ++i) {
                    float f = v[i];
                    bf16 hi, lo; split_bf16(f, hi, lo);
                    eh[(d + i) * FEP + code] = hi;
                    el[(d + i) * FEP + code] = lo;
                    ssum = fmaf(f, f, ssum);
                }
            }
            ssum += __shfl_xor(ssum, 1);
            if (h == 0) esq[code] = 0.5f * ssum;
        }
        __syncthreads();

        v8f acc[4];
        #pragma unroll
        for (int ct = 0; ct < 4; ++ct)
            acc[ct] = (v8f){0.f,0.f,0.f,0.f,0.f,0.f,0.f,0.f};

        const int rowl  = wave * 16 + (lane & 15);
        const int khalf = lane >> 4;

        #pragma unroll
        for (int s = 0; s < 8; ++s) {
            const int d0 = 32 * s + khalf * 8;
            Frag Ah, Al;
            const u32x4* pah = (const u32x4*)(xh + rowl * XPITCH + d0);
            const u32x4* pal = (const u32x4*)(xl + rowl * XPITCH + d0);
            Ah.q[0] = pah[0]; Ah.q[1] = pah[2];
            Al.q[0] = pal[0]; Al.q[1] = pal[2];
            const int dB = 32 * s + khalf * 8;
            #pragma unroll
            for (int ct = 0; ct < 4; ++ct) {
                Frag Bh, Bl;
                #pragma unroll
                for (int e = 0; e < 16; ++e) {
                    const int kk = dB + (e & 7) + (e >> 3) * 16;
                    Bh.v[e] = eh[kk * FEP + ct * 16 + (lane & 15)];
                    Bl.v[e] = el[kk * FEP + ct * 16 + (lane & 15)];
                }
                acc[ct] = wmma_bf16(Ah.v, Bh.v, acc[ct]);
                acc[ct] = wmma_bf16(Ah.v, Bl.v, acc[ct]);
                acc[ct] = wmma_bf16(Al.v, Bh.v, acc[ct]);
            }
        }

        #pragma unroll
        for (int ct = 0; ct < 4; ++ct) {
            const int nl = ct * 16 + (lane & 15);
            const float hs = esq[nl];
            const int kg = kbase + nl;
            #pragma unroll
            for (int r = 0; r < 8; ++r) {
                const float sc = acc[ct][r] - hs;
                if (sc > bestv[r]) { bestv[r] = sc; besti[r] = kg; }
            }
        }
    }

    #pragma unroll
    for (int r = 0; r < 8; ++r) {
        float v = bestv[r];
        int   i = besti[r];
        #pragma unroll
        for (int m = 1; m < 16; m <<= 1) {
            const float ov = __shfl_xor(v, m);
            const int   oi = __shfl_xor(i, m);
            if (ov > v || (ov == v && oi < i)) { v = ov; i = oi; }
        }
        if ((lane & 15) == 0)
            sidx[wave * 16 + (lane >> 4) * 8 + r] = i;
    }
    __syncthreads();

    if (t < TILE_M) out_idx[row0 + t] = sidx[t];
    {
        const int r = t >> 1;
        const int h = t & 1;
        const int k = sidx[r];
        const f32x4* sp = (const f32x4*)(embed + (size_t)k * DIM + h * 128);
        f32x4*       dp = (f32x4*)(out_q + (size_t)(row0 + r) * DIM + h * 128);
        #pragma unroll
        for (int j = 0; j < 32; ++j) dp[j] = sp[j];
    }
}

extern "C" void kernel_launch(void* const* d_in, const int* in_sizes, int n_in,
                              void* d_out, int out_size, void* d_ws, size_t ws_size,
                              hipStream_t stream) {
    (void)n_in; (void)out_size;
    const float* x     = (const float*)d_in[0];
    const float* embed = (const float*)d_in[1];
    const int N = in_sizes[0] / DIM;   // 16384
    const int K = in_sizes[1] / DIM;   // 8192

    float* out_q   = (float*)d_out;
    int*   out_idx = (int*)(out_q + (size_t)N * DIM);

    const int nChunks = K / CH;
    const size_t ws_needed = (size_t)nChunks * PANEL_BYTES + (size_t)K * sizeof(float);

    dim3 block(128);
    dim3 grid(N / TILE_M);

    if (d_ws != nullptr && ws_size >= ws_needed) {
        bf16*  panels = (bf16*)d_ws;
        float* esq_g  = (float*)((char*)d_ws + (size_t)nChunks * PANEL_BYTES);
        vq_pre<<<dim3(nChunks), dim3(256), 0, stream>>>(embed, panels, esq_g);
        vq_main<<<grid, block, MAIN_SMEM, stream>>>(x, panels, esq_g, embed,
                                                    out_q, out_idx, K);
    } else {
        vq_fallback<<<grid, block, FB_SMEM, stream>>>(x, embed, out_q, out_idx, K);
    }
}
